// TestModel_38680475467916
// MI455X (gfx1250) — 8-bit matrix operand order repaired and checked by reading, compiled, not run on hardware
//
#include <hip/hip_runtime.h>
#include <hip/hip_bf16.h>

constexpr int H = 4096;
constexpr int T = 16384;
constexpr float FP8_MAX = 448.0f;
constexpr float RMS_EPS = 1e-6f;

typedef __attribute__((ext_vector_type(16))) int   v16i;
typedef __attribute__((ext_vector_type(8)))  float v8f;

#if defined(__has_builtin)
#if __has_builtin(__builtin_amdgcn_cvt_pk_fp8_f32)
#define USE_HW_FP8_CVT 1
#endif
#endif

__device__ __forceinline__ unsigned int f32_to_e4m3_manual(float f) {
  unsigned int u    = __float_as_uint(f);
  unsigned int sign = (u >> 31) << 7;
  unsigned int a    = u & 0x7FFFFFFFu;
  if (a >= 0x3C800000u) {
    unsigned int mant = a & 0x7FFFFFu;
    int e             = (int)(a >> 23) - 127;
    unsigned int m3   = mant >> 20;
    unsigned int rem  = mant & 0xFFFFFu;
    if (rem > 0x80000u || (rem == 0x80000u && (m3 & 1u))) ++m3;
    unsigned int ex = (unsigned int)(e + 7);
    if (m3 == 8u) { m3 = 0u; ++ex; }
    if (ex > 15u || (ex == 15u && m3 > 6u)) { ex = 15u; m3 = 6u; }
    return sign | (ex << 3) | m3;
  } else {
    int m = __float2int_rn(fabsf(f) * 512.0f);
    return sign | (unsigned int)m;
  }
}

__device__ __forceinline__ unsigned int f32_to_e4m3(float f) {
#ifdef USE_HW_FP8_CVT
  return (unsigned int)(__builtin_amdgcn_cvt_pk_fp8_f32(f, 0.0f, 0, false) & 0xFF);
#else
  return f32_to_e4m3_manual(f);
#endif
}

__global__ __launch_bounds__(256) void quant_weight_t(
    const float* __restrict__ wt, unsigned char* __restrict__ wq) {
  __shared__ unsigned char tile[32][33];
  const int n0 = blockIdx.x * 32;
  const int k0 = blockIdx.y * 32;
  const int tx = threadIdx.x & 31;
  const int ty = threadIdx.x >> 5;
#pragma unroll
  for (int i = 0; i < 4; ++i) {
    int k = ty + i * 8;
    float v = wt[(size_t)(k0 + k) * H + (n0 + tx)];
    tile[k][tx] = (unsigned char)f32_to_e4m3(v);
  }
  __syncthreads();
#pragma unroll
  for (int i = 0; i < 4; ++i) {
    int n = ty + i * 8;
    wq[(size_t)(n0 + n) * H + (k0 + tx)] = tile[tx][n];
  }
}

template <int MODE>
__global__ __launch_bounds__(256) void rms_quant(
    const float* __restrict__ src, const float* __restrict__ nw,
    unsigned char* __restrict__ q, float* __restrict__ srow,
    float* __restrict__ resid, float* __restrict__ out) {
  __shared__ float red[256];
  const int t   = blockIdx.x;
  const int tid = threadIdx.x;
  const float4* row  = (const float4*)(src + (size_t)t * H);
  const float4* nwv4 = (const float4*)nw;

  float4 xv[4], wv[4];
  float ss = 0.0f;
#pragma unroll
  for (int j = 0; j < 4; ++j) {
    int f = tid + 256 * j;
    xv[j] = row[f];
    wv[j] = nwv4[f];
    ss += xv[j].x * xv[j].x + xv[j].y * xv[j].y +
          xv[j].z * xv[j].z + xv[j].w * xv[j].w;
  }
  if (MODE == 0) {
    float4* rr = (float4*)(resid + (size_t)t * H);
#pragma unroll
    for (int j = 0; j < 4; ++j) {
      int f = tid + 256 * j;
      float4 r;
      r.x = sqrtf(xv[j].x); r.y = sqrtf(xv[j].y);
      r.z = sqrtf(xv[j].z); r.w = sqrtf(xv[j].w);
      rr[f] = r;
    }
  }

  red[tid] = ss;
  __syncthreads();
  for (int s = 128; s > 0; s >>= 1) {
    if (tid < s) red[tid] += red[tid + s];
    __syncthreads();
  }
  const float rrms = rsqrtf(red[0] / (float)H + RMS_EPS);
  __syncthreads();

  float amax = 0.0f;
#pragma unroll
  for (int j = 0; j < 4; ++j) {
    xv[j].x *= rrms * wv[j].x; xv[j].y *= rrms * wv[j].y;
    xv[j].z *= rrms * wv[j].z; xv[j].w *= rrms * wv[j].w;
    amax = fmaxf(amax, fmaxf(fmaxf(fabsf(xv[j].x), fabsf(xv[j].y)),
                             fmaxf(fabsf(xv[j].z), fabsf(xv[j].w))));
  }

  if (MODE == 2) {
    float4* o = (float4*)(out + (size_t)t * H);
#pragma unroll
    for (int j = 0; j < 4; ++j) o[tid + 256 * j] = xv[j];
    return;
  }

  red[tid] = amax;
  __syncthreads();
  for (int s = 128; s > 0; s >>= 1) {
    if (tid < s) red[tid] = fmaxf(red[tid], red[tid + s]);
    __syncthreads();
  }
  const float sc  = fmaxf(red[0] / FP8_MAX, 1e-10f);
  const float inv = 1.0f / sc;
  if (tid == 0) srow[t] = sc;

  unsigned int* qrow = (unsigned int*)(q + (size_t)t * H);
#pragma unroll
  for (int j = 0; j < 4; ++j) {
    float q0 = fminf(fmaxf(xv[j].x * inv, -FP8_MAX), FP8_MAX);
    float q1 = fminf(fmaxf(xv[j].y * inv, -FP8_MAX), FP8_MAX);
    float q2 = fminf(fmaxf(xv[j].z * inv, -FP8_MAX), FP8_MAX);
    float q3 = fminf(fmaxf(xv[j].w * inv, -FP8_MAX), FP8_MAX);
#ifdef USE_HW_FP8_CVT
    int lo = __builtin_amdgcn_cvt_pk_fp8_f32(q0, q1, 0, false);
    unsigned int packed =
        (unsigned int)__builtin_amdgcn_cvt_pk_fp8_f32(q2, q3, lo, true);
#else
    unsigned int packed = f32_to_e4m3(q0) | (f32_to_e4m3(q1) << 8) |
                          (f32_to_e4m3(q2) << 16) | (f32_to_e4m3(q3) << 24);
#endif
    qrow[tid + 256 * j] = packed;
  }
}

__device__ __forceinline__ void async_ld16(unsigned lds, unsigned voff,
                                           unsigned long long base) {
  asm volatile("global_load_async_to_lds_b128 %0, %1, %2"
               :: "v"(lds), "v"(voff), "s"(base) : "memory");
}

constexpr int LDS_STRIDE = 144;
constexpr int TILE_BYTES = 128 * LDS_STRIDE;
constexpr int STAGE_BYTES = 2 * TILE_BYTES;

__global__ __launch_bounds__(256) void gemm_fp8_resid(
    const unsigned char* __restrict__ Aq, const unsigned char* __restrict__ Bq,
    const float* __restrict__ srow, const float* __restrict__ wscale,
    float* __restrict__ resid) {
  __shared__ __align__(16) unsigned char smem[2 * STAGE_BYTES];

  const int tid  = threadIdx.x;
  const int lane = tid & 31;
  const int wave = tid >> 5;
  const int wm   = wave >> 1;
  const int wn   = wave & 1;
  const int bm   = blockIdx.y * 128;
  const int bn   = blockIdx.x * 128;
  const int ml   = lane & 15;
  const int half = lane >> 4;

  const int rowf = tid >> 3;
  const int colf = (tid & 7) * 16;
  const unsigned smbase = (unsigned)(uintptr_t)&smem[0];
  unsigned voffA[4], voffB[4], ldsA[4], ldsB[4];
#pragma unroll
  for (int i = 0; i < 4; ++i) {
    voffA[i] = (unsigned)((bm + rowf + 32 * i) * H + colf);
    voffB[i] = (unsigned)((bn + rowf + 32 * i) * H + colf);
    ldsA[i]  = smbase + (rowf + 32 * i) * LDS_STRIDE + colf;
    ldsB[i]  = ldsA[i] + TILE_BYTES;
  }
  const unsigned long long Abase = (unsigned long long)(uintptr_t)Aq;
  const unsigned long long Bbase = (unsigned long long)(uintptr_t)Bq;

  const int aoff = (wm * 32 + ml) * LDS_STRIDE + half * 16;
  const int boff = (wn * 64 + ml) * LDS_STRIDE + half * 16;

  v8f acc[2][4] = {};

#pragma unroll
  for (int i = 0; i < 4; ++i) async_ld16(ldsA[i], voffA[i], Abase);
#pragma unroll
  for (int i = 0; i < 4; ++i) async_ld16(ldsB[i], voffB[i], Bbase);

  constexpr int NK = H / 128;
  unsigned long long kload = 128;
  for (int k = 0; k < NK; ++k) {
    const int s = k & 1;
    if (k + 1 < NK) {
      const unsigned so = (unsigned)((s ^ 1) * STAGE_BYTES);
      const unsigned long long ab = Abase + kload;
      const unsigned long long bb = Bbase + kload;
#pragma unroll
      for (int i = 0; i < 4; ++i) async_ld16(ldsA[i] + so, voffA[i], ab);
#pragma unroll
      for (int i = 0; i < 4; ++i) async_ld16(ldsB[i] + so, voffB[i], bb);
      kload += 128;
      asm volatile("s_wait_asynccnt 0x8" ::: "memory");
    } else {
      asm volatile("s_wait_asynccnt 0x0" ::: "memory");
    }
    __syncthreads();

    const unsigned char* As = smem + s * STAGE_BYTES;
    const unsigned char* Bs = As + TILE_BYTES;

    v16i a0, a1, b[4];
    const unsigned char* ap0 = As + aoff;
    const unsigned char* ap1 = ap0 + 16 * LDS_STRIDE;
#pragma unroll
    for (int c = 0; c < 8; ++c) {
      int2 d0 = *(const int2*)(ap0 + 32 * (c >> 1) + 8 * (c & 1));
      int2 d1 = *(const int2*)(ap1 + 32 * (c >> 1) + 8 * (c & 1));
      a0[2 * c] = d0.x; a0[2 * c + 1] = d0.y;
      a1[2 * c] = d1.x; a1[2 * c + 1] = d1.y;
    }
#pragma unroll
    for (int nt = 0; nt < 4; ++nt) {
      const unsigned char* bp = Bs + boff + nt * 16 * LDS_STRIDE;
#pragma unroll
      for (int g = 0; g < 4; ++g) {
        int4 d = *(const int4*)(bp + 32 * g);
        b[nt][4 * g]     = d.x; b[nt][4 * g + 1] = d.y;
        b[nt][4 * g + 2] = d.z; b[nt][4 * g + 3] = d.w;
      }
    }
#pragma unroll
    for (int nt = 0; nt < 4; ++nt) {
      acc[0][nt] = __builtin_amdgcn_wmma_f32_16x16x128_fp8_fp8(
          a0, b[nt], (short)0, acc[0][nt], false, false);
      acc[1][nt] = __builtin_amdgcn_wmma_f32_16x16x128_fp8_fp8(
          a1, b[nt], (short)0, acc[1][nt], false, false);
    }
    __syncthreads();
  }

#pragma unroll
  for (int mt = 0; mt < 2; ++mt) {
#pragma unroll
    for (int nt = 0; nt < 4; ++nt) {
      const int col   = bn + wn * 64 + nt * 16 + ml;
      const float wsc = wscale[col];
#pragma unroll
      for (int r = 0; r < 8; ++r) {
        const int row = bm + wm * 32 + mt * 16 + half * 8 + r;
        const size_t idx = (size_t)row * H + col;
        resid[idx] += acc[mt][nt][r] * srow[row] * wsc;
      }
    }
  }
}

extern "C" void kernel_launch(void* const* d_in, const int* in_sizes, int n_in,
                              void* d_out, int out_size, void* d_ws, size_t ws_size,
                              hipStream_t stream) {
  (void)in_sizes; (void)n_in; (void)out_size; (void)ws_size;
  const float* x   = (const float*)d_in[0];
  const float* nw0 = (const float*)d_in[1];
  const float* nw1 = (const float*)d_in[2];
  const float* nw2 = (const float*)d_in[3];
  const float* w0  = (const float*)d_in[4];
  const float* w1  = (const float*)d_in[5];
  const float* s0  = (const float*)d_in[6];
  const float* s1  = (const float*)d_in[7];
  float* out = (float*)d_out;

  char* ws = (char*)d_ws;
  float*         resid = (float*)ws;
  unsigned char* aq    = (unsigned char*)(ws + (size_t)T * H * 4);
  unsigned char* wq0   = aq  + (size_t)T * H;
  unsigned char* wq1   = wq0 + (size_t)H * H;
  float*         srow  = (float*)(wq1 + (size_t)H * H);

  const dim3 gW(H / 32, H / 32);
  const dim3 gG(H / 128, T / 128);

  quant_weight_t<<<gW, 256, 0, stream>>>(w0, wq0);
  quant_weight_t<<<gW, 256, 0, stream>>>(w1, wq1);

  rms_quant<0><<<T, 256, 0, stream>>>(x, nw0, aq, srow, resid, nullptr);
  gemm_fp8_resid<<<gG, 256, 0, stream>>>(aq, wq0, srow, s0, resid);

  rms_quant<1><<<T, 256, 0, stream>>>(resid, nw1, aq, srow, nullptr, nullptr);
  gemm_fp8_resid<<<gG, 256, 0, stream>>>(aq, wq1, srow, s1, resid);

  rms_quant<2><<<T, 256, 0, stream>>>(resid, nw2, nullptr, nullptr, nullptr, out);
}
